// ShiftedWindowAttn3_82205674046067
// MI455X (gfx1250) — matrix operand order repaired and checked by reading, not compiled, not run on hardware
//
#include <hip/hip_runtime.h>

typedef _Float16 f16;
typedef _Float16 v16h __attribute__((ext_vector_type(16)));
typedef _Float16 v8h  __attribute__((ext_vector_type(8)));
typedef float    v8f  __attribute__((ext_vector_type(8)));

#define DEV static __device__ __forceinline__

constexpr int C    = 96;    // channels
constexpr int NH   = 3;     // heads
constexpr int HD   = 32;    // head dim
constexpr int WSZ  = 7;     // window size
constexpr int LTOK = 49;    // tokens per window
constexpr int LPAD = 64;    // padded tokens
constexpr int IMG  = 112;   // image H=W
constexpr int SH   = 3;     // shift
constexpr float QSCALE = 0.17677669529663689f; // 1/sqrt(32)

// ---- WMMA fragment helpers (contiguous k order in both A and B, wave32) ----------

// A fragment: 16x32 f16, row m = m0 + lane%16.
// half=lane/16: VGPR0-7 hold K = k0+half*16 .. +16 consecutive
DEV v16h load_a(const f16* mat, int stride, int m0, int k0, int lane) {
  int m    = m0 + (lane & 15);
  int half = lane >> 4;
  const f16* row = mat + m * stride;
  v8h lo = *(const v8h*)(row + k0 + half * 16);       // 16B aligned
  v8h hi = *(const v8h*)(row + k0 + half * 16 + 8);   // 16B aligned
  v16h a;
#pragma unroll
  for (int j = 0; j < 8; ++j) { a[j] = lo[j]; a[8 + j] = hi[j]; }
  return a;
}

// B fragment for B[k][n] = src[n][k] (src row-major [N][K]):
// lane holds column n = n0 + lane%16, K range k0 + (lane/16)*16 .. +16 consecutive.
DEV v16h load_bt(const f16* mat, int stride, int n0, int k0, int lane) {
  int n    = n0 + (lane & 15);
  int half = lane >> 4;
  return *(const v16h*)(mat + n * stride + k0 + half * 16);  // 32B aligned at all call sites
}

DEV v8f wmma_f16(v16h a, v16h b, v8f c) {
  return __builtin_amdgcn_wmma_f32_16x16x32_f16(false, a, false, b, (short)0, c, false, false);
}

// ---- weight f32 -> f16 pre-conversion --------------------------------------

__global__ void convert_weights(const float* __restrict__ qw, const float* __restrict__ pw,
                                f16* __restrict__ wq16, f16* __restrict__ wp16) {
  int i = blockIdx.x * blockDim.x + threadIdx.x;
  if (i < 3 * C * C) wq16[i] = (f16)qw[i];
  if (i < C * C)     wp16[i] = (f16)pw[i];
}

// ---- main: one workgroup (8 wave32) per shifted window ---------------------
// All "token-output" GEMMs are computed transposed (D = W x Tokens^T) so each
// lane's 8 D values are channel-consecutive -> single b128 stores.

__global__ __launch_bounds__(256) void swin_attn_kernel(
    const float* __restrict__ x, const float* __restrict__ qkv_bias,
    const float* __restrict__ proj_bias, const float* __restrict__ rel_bias,
    const f16* __restrict__ wq, const f16* __restrict__ wp,
    float* __restrict__ out)
{
  extern __shared__ char smem[];
  f16*   sX   = (f16*)smem;              // [64][96] shifted window input
  f16*   sQ   = sX  + LPAD * C;          // [64][96] scaled Q (all heads)
  f16*   sK   = sQ  + LPAD * C;          // [64][96] K (all heads)
  f16*   sVt  = sK  + LPAD * C;          // [96][64] V transposed (row = channel)
  f16*   sO   = sVt + C * LPAD;          // [64][96] attention output (heads concat)
  f16*   sP   = sO  + LPAD * C;          // [64][64] softmax probs f16
  float* sS   = (float*)(sP + LPAD * LPAD); // [64][64] scores f32
  int*   sLab = (int*)(sS + LPAD * LPAD);   // [64] region labels
  int*   sOff = sLab + LPAD;                // [64] output base offsets (-1 = pad row)

  int blk  = blockIdx.x;
  int img  = blk >> 8;
  int win  = blk & 255;
  int wh   = win >> 4, ww = win & 15;
  int tid  = threadIdx.x, lane = tid & 31, wave = tid >> 5;

  // ---- per-token labels + global offsets (roll by +shift both directions) -
  if (tid < LPAD) {
    int lab = 0, off = -1;
    if (tid < LTOK) {
      int i = tid / WSZ, j = tid % WSZ;
      int r  = wh * WSZ + i, c = ww * WSZ + j;          // shifted coords
      int lh = (r < IMG - WSZ) ? 0 : ((r < IMG - SH) ? 1 : 2);
      int lw = (c < IMG - WSZ) ? 0 : ((c < IMG - SH) ? 1 : 2);
      lab = lh * 3 + lw;
      int rr = (r + SH) % IMG, cc = (c + SH) % IMG;     // original coords
      off = ((img * IMG + rr) * IMG + cc) * C;
    }
    sLab[tid] = lab;
    sOff[tid] = off;
  }

  // ---- stage X: one token x 24 channels per thread, vectorized ------------
  {
    int t  = tid >> 2;              // 0..63
    int qc = (tid & 3) * 24;        // 0,24,48,72
    f16* dstp = sX + t * C + qc;
    if (t < LTOK) {
      int i = t / WSZ, j = t % WSZ;
      int r = (wh * WSZ + i + SH) % IMG;
      int c = (ww * WSZ + j + SH) % IMG;
      const float* src = x + ((img * IMG + r) * IMG + c) * C + qc;
#pragma unroll
      for (int v = 0; v < 3; ++v) {
        float4 a = *(const float4*)(src + v * 8);
        float4 b = *(const float4*)(src + v * 8 + 4);
        v8h h8;
        h8[0]=(f16)a.x; h8[1]=(f16)a.y; h8[2]=(f16)a.z; h8[3]=(f16)a.w;
        h8[4]=(f16)b.x; h8[5]=(f16)b.y; h8[6]=(f16)b.z; h8[7]=(f16)b.w;
        *(v8h*)(dstp + v * 8) = h8;
      }
    } else {
      v8h z = {};
#pragma unroll
      for (int v = 0; v < 3; ++v) *(v8h*)(dstp + v * 8) = z;
    }
  }
  __syncthreads();

  // ---- GEMM1: QKV^T = Wqkv[288x96] @ X^T -> 18x4 tiles, 9 per wave --------
  // D[n][m]: lane owns token m = m0+lane%16, rows n = n0 + g + 8*half.
  for (int it = 0; it < 9; ++it) {
    int t  = wave * 9 + it;
    int mt = t & 3, nt = t >> 2;       // mt: token tile, nt: W-row tile (0..17)
    int m0 = mt * 16, n0 = nt * 16;
    v8f acc = {};
#pragma unroll
    for (int kk = 0; kk < 3; ++kk) {
      v16h a = load_a (wq, C, n0, kk * 32, lane);
      v16h b = load_bt(sX, C, m0, kk * 32, lane);
      acc = wmma_f16(a, b, acc);
    }
    int m    = m0 + (lane & 15);
    int half = lane >> 4;
    int nb   = n0 + 8 * half;          // first of 8 consecutive channels
    float4 b0 = *(const float4*)(qkv_bias + nb);
    float4 b1 = *(const float4*)(qkv_bias + nb + 4);
    float bias[8] = {b0.x, b0.y, b0.z, b0.w, b1.x, b1.y, b1.z, b1.w};
    if (nt < 6) {                      // Q, scaled: one b128 store
      v8h hv;
#pragma unroll
      for (int g = 0; g < 8; ++g) hv[g] = (f16)((acc[g] + bias[g]) * QSCALE);
      *(v8h*)(sQ + m * C + nb) = hv;
    } else if (nt < 12) {              // K: one b128 store
      v8h hv;
#pragma unroll
      for (int g = 0; g < 8; ++g) hv[g] = (f16)(acc[g] + bias[g]);
      *(v8h*)(sK + m * C + (nb - C)) = hv;
    } else {                           // V, transposed (row = channel)
      f16* col = sVt + (nb - 2 * C) * LPAD + m;
#pragma unroll
      for (int g = 0; g < 8; ++g) col[g * LPAD] = (f16)(acc[g] + bias[g]);
    }
  }
  __syncthreads();

  // ---- per head: S = Q K^T, softmax(+bias+mask), O = P V ------------------
  for (int h = 0; h < NH; ++h) {
    // S: 4x4 tiles, K=32 (single WMMA), 2 tiles per wave; raw scores only
    for (int it = 0; it < 2; ++it) {
      int t  = wave * 2 + it;
      int mt = t >> 2, nt = t & 3;
      int m0 = mt * 16, n0 = nt * 16;
      v16h a = load_a (sQ, C, m0, h * HD, lane);
      v16h b = load_bt(sK, C, n0, h * HD, lane);
      v8f acc = {};
      acc = wmma_f16(a, b, acc);
      int n    = n0 + (lane & 15);
      int half = lane >> 4;
      float* dst = sS + (m0 + 8 * half) * LPAD + n;   // lanes 0-15 consecutive
#pragma unroll
      for (int g = 0; g < 8; ++g) dst[g * LPAD] = acc[g];
    }
    __syncthreads();

    // softmax: 4 threads per row, bias+mask folded into pass 1
    {
      int row = tid >> 2;            // 0..63
      int sub = tid & 3;
      float* srow = sS + row * LPAD;
      f16*   prow = sP + row * LPAD;
      int    labm = sLab[row];
      const float* rb = rel_bias + (h * LTOK + row) * LTOK;
      bool valid = row < LTOK;
      float mx = -1e30f;
      for (int k = sub; k < LTOK; k += 4) {
        float v = srow[k];
        if (valid) {
          v += rb[k];
          if (labm != sLab[k]) v -= 100.f;
        }
        srow[k] = v;                 // each element owned by exactly one thread
        mx = fmaxf(mx, v);
      }
      mx = fmaxf(mx, __shfl_xor(mx, 1, 32));
      mx = fmaxf(mx, __shfl_xor(mx, 2, 32));
      float sum = 0.f;
      for (int k = sub; k < LTOK; k += 4) {
        float e = __expf(srow[k] - mx);
        sum += e;
        prow[k] = (f16)e;
      }
      sum += __shfl_xor(sum, 1, 32);
      sum += __shfl_xor(sum, 2, 32);
      float inv = 1.f / sum;
      for (int k = sub; k < LTOK; k += 4) prow[k] = (f16)((float)prow[k] * inv);
      for (int k = LTOK + sub; k < LPAD; k += 4) prow[k] = (f16)0.f;
    }
    __syncthreads();

    // O^T = V^T[32x64] @ P^T: 2x4 tiles, 1 per wave, K=64 (2 WMMA)
    // D[n][m]: lane owns token m, rows = head-local channels -> b128 store
    {
      int ntl = wave & 1, mt = wave >> 1;
      int n0 = ntl * 16, m0 = mt * 16;
      v8f acc = {};
#pragma unroll
      for (int kk = 0; kk < 2; ++kk) {
        v16h a = load_a (sVt, LPAD, h * HD + n0, kk * 32, lane);
        v16h b = load_bt(sP,  LPAD, m0,          kk * 32, lane);
        acc = wmma_f16(a, b, acc);
      }
      int m    = m0 + (lane & 15);
      int half = lane >> 4;
      int nb   = h * HD + n0 + 8 * half;
      v8h hv;
#pragma unroll
      for (int g = 0; g < 8; ++g) hv[g] = (f16)acc[g];
      *(v8h*)(sO + m * C + nb) = hv;
    }
    __syncthreads();
  }

  // ---- GEMM4: Y^T = Wproj[96x96] @ O^T + b -> 6x4 tiles, 3 per wave -------
  // lane owns token m: one offset lookup + one branch; 2x b128 global stores.
  for (int it = 0; it < 3; ++it) {
    int t  = wave * 3 + it;
    int mt = t & 3, nt = t >> 2;       // nt < 6
    int m0 = mt * 16, n0 = nt * 16;
    v8f acc = {};
#pragma unroll
    for (int kk = 0; kk < 3; ++kk) {
      v16h a = load_a (wp, C, n0, kk * 32, lane);
      v16h b = load_bt(sO, C, m0, kk * 32, lane);
      acc = wmma_f16(a, b, acc);
    }
    int m    = m0 + (lane & 15);
    int half = lane >> 4;
    int nb   = n0 + 8 * half;
    int off  = sOff[m];
    if (off >= 0) {
      float4 b0 = *(const float4*)(proj_bias + nb);
      float4 b1 = *(const float4*)(proj_bias + nb + 4);
      float4 o0, o1;
      o0.x = acc[0] + b0.x; o0.y = acc[1] + b0.y; o0.z = acc[2] + b0.z; o0.w = acc[3] + b0.w;
      o1.x = acc[4] + b1.x; o1.y = acc[5] + b1.y; o1.z = acc[6] + b1.z; o1.w = acc[7] + b1.w;
      *(float4*)(out + off + nb)     = o0;
      *(float4*)(out + off + nb + 4) = o1;
    }
  }
}

// ---- host launch -----------------------------------------------------------

extern "C" void kernel_launch(void* const* d_in, const int* in_sizes, int n_in,
                              void* d_out, int out_size, void* d_ws, size_t ws_size,
                              hipStream_t stream) {
  (void)in_sizes; (void)n_in; (void)out_size; (void)ws_size;
  const float* x      = (const float*)d_in[0];
  const float* qkv_w  = (const float*)d_in[1];
  const float* qkv_b  = (const float*)d_in[2];
  const float* proj_w = (const float*)d_in[3];
  const float* proj_b = (const float*)d_in[4];
  const float* rel_b  = (const float*)d_in[5];
  float* out = (float*)d_out;

  f16* wq16 = (f16*)d_ws;                  // 288*96 f16
  f16* wp16 = wq16 + 3 * C * C;            // 96*96 f16

  convert_weights<<<(3 * C * C + 255) / 256, 256, 0, stream>>>(qkv_w, proj_w, wq16, wp16);

  size_t smem = (size_t)(5 * LPAD * C + LPAD * LPAD) * sizeof(f16) // X,Q,K,Vt,O,P
              + (size_t)(LPAD * LPAD) * sizeof(float)              // S
              + (size_t)(2 * LPAD) * sizeof(int);                  // labels + offsets
  dim3 grid(32 * 256);                     // one block per window
  swin_attn_kernel<<<grid, 256, smem, stream>>>(x, qkv_b, proj_b, rel_b, wq16, wp16, out);
}
